// DAxialConv2d_45973329936721
// MI455X (gfx1250) — compile-verified, matrix operand order checked by reading, not run on hardware
//
#include <hip/hip_runtime.h>
#include <hip/hip_bf16.h>

// ---------------------------------------------------------------------------
// DAxialConv2d for MI455X (gfx1250, wave32, WMMA 16x16x32 f16->f32)
// ---------------------------------------------------------------------------
typedef _Float16 v16h __attribute__((ext_vector_type(16)));
typedef float    v8f  __attribute__((ext_vector_type(8)));

#define Bn 8
#define Cc 64
#define Hh 128
#define Ww 128
#define HWp (Hh*Ww)

#define XH_ELEMS    ((size_t)Bn * Cc * Hh * Ww)   // 8,388,608 f32
#define NKHW        ((size_t)Bn * 3 * Hh * Ww)    // 393,216 f32 per coord array
#define WFRAG_HALFS (4 * 6 * 32 * 16)             // 12,288 f16 per pass

// ---------------------------------------------------------------------------
// Kernel 1: swizzle conv weights (o,c,tap) -> f16 WMMA A-fragments.
// K_lin = tap*64 + c ; chunk t = K_lin>>5 ; within-chunk kk = K_lin&31.
// ISA 16-bit A 16x32 layout: lane = M (lanes>=16 add 8 to kk's lane-half);
// half j: kk = (j&8)*2 + (j&7) + (lane>=16 ? 8 : 0).
// ---------------------------------------------------------------------------
__global__ void prep_weights_kernel(const float* __restrict__ w_h,
                                    const float* __restrict__ w_v,
                                    _Float16* __restrict__ wfrag) {
  int idx = blockIdx.x * blockDim.x + threadIdx.x;
  if (idx >= 2 * WFRAG_HALFS) return;
  int pass = idx / WFRAG_HALFS;
  int r    = idx % WFRAG_HALFS;
  int j    = r & 15;
  int lane = (r >> 4) & 31;
  int t    = (r >> 9) % 6;
  int ot   = (r >> 9) / 6;
  int kk    = ((j & 8) << 1) + (j & 7) + ((lane >= 16) ? 8 : 0);
  int K_lin = t * 32 + kk;
  int tap   = K_lin >> 6;       // K_lin = tap*64 + c
  int c     = K_lin & 63;
  int o     = ot * 16 + (lane & 15);
  const float* w = pass ? w_v : w_h;   // both flatten as o*192 + c*3 + tap
  wfrag[idx] = (_Float16)w[o * 192 + c * 3 + tap];
}

// ---------------------------------------------------------------------------
// Kernel 2: offset/mask convs for both passes; emit final sample coords.
// taps = [-1,0,1] (K=3, DIL=1, PAD=1). Zero padding at borders.
// ---------------------------------------------------------------------------
__global__ void offsets_masks_kernel(const float* __restrict__ X,
    const float* __restrict__ Woh, const float* __restrict__ Boh,
    const float* __restrict__ Wmh, const float* __restrict__ Bmh,
    const float* __restrict__ Wov, const float* __restrict__ Bov,
    const float* __restrict__ Wmv, const float* __restrict__ Bmv,
    float* __restrict__ PYH, float* __restrict__ PXH, float* __restrict__ MH,
    float* __restrict__ PYV, float* __restrict__ PXV, float* __restrict__ MV) {
  int idx = blockIdx.x * blockDim.x + threadIdx.x;
  if (idx >= Bn * Hh * Ww) return;
  int w = idx % Ww;
  int h = (idx / Ww) % Hh;
  int n = idx / (Hh * Ww);

  float offh[6] = {}, mh[3] = {}, offv[6] = {}, mv[3] = {};
  const float* xb = X + (size_t)n * Cc * HWp + h * Ww + w;
  for (int c = 0; c < Cc; ++c) {
    const float* xc = xb + (size_t)c * HWp;
    float x0 = xc[0];
    float xl = (w > 0)      ? xc[-1]  : 0.f;
    float xr = (w < Ww - 1) ? xc[1]   : 0.f;
    float xu = (h > 0)      ? xc[-Ww] : 0.f;
    float xd = (h < Hh - 1) ? xc[Ww]  : 0.f;
    int cb = c * 3;
#pragma unroll
    for (int o = 0; o < 6; ++o) {
      offh[o] += Woh[o*192+cb]*xl + Woh[o*192+cb+1]*x0 + Woh[o*192+cb+2]*xr;
      offv[o] += Wov[o*192+cb]*xu + Wov[o*192+cb+1]*x0 + Wov[o*192+cb+2]*xd;
    }
#pragma unroll
    for (int o = 0; o < 3; ++o) {
      mh[o] += Wmh[o*192+cb]*xl + Wmh[o*192+cb+1]*x0 + Wmh[o*192+cb+2]*xr;
      mv[o] += Wmv[o*192+cb]*xu + Wmv[o*192+cb+1]*x0 + Wmv[o*192+cb+2]*xd;
    }
  }
#pragma unroll
  for (int k = 0; k < 3; ++k) {
    size_t o3 = (((size_t)n * 3 + k) * Hh + h) * Ww + w;
    PYH[o3] = (float)h + offh[2*k] + Boh[2*k];
    PXH[o3] = (float)w + (float)(k - 1) + offh[2*k+1] + Boh[2*k+1];
    MH[o3]  = 1.f / (1.f + __expf(-(mh[k] + Bmh[k])));
    PYV[o3] = (float)h + (float)(k - 1) + offv[2*k] + Bov[2*k];
    PXV[o3] = (float)w + offv[2*k+1] + Bov[2*k+1];
    MV[o3]  = 1.f / (1.f + __expf(-(mv[k] + Bmv[k])));
  }
}

// ---------------------------------------------------------------------------
// Kernel 3: deformable axial pass. 4 waves/block, 16 pixels/wave, all 64
// output channels. Phase 1: masked bilinear gather -> f16 B-fragments in LDS.
// Phase 2: 6 x (K=32) chunks x 4 M-tiles of v_wmma_f32_16x16x32_f16.
// ---------------------------------------------------------------------------
__global__ __launch_bounds__(128) void deform_pass_kernel(
    const float* __restrict__ src,
    const float* __restrict__ PY, const float* __restrict__ PX,
    const float* __restrict__ M,
    const _Float16* __restrict__ wfrag,
    const float* __restrict__ bias,
    float* __restrict__ dst) {
  __shared__ __align__(16) _Float16 smem[4 * 6 * 32 * 16];  // 4 waves * 3072 halfs
  int lane = threadIdx.x & 31;
  int wave = threadIdx.x >> 5;
  int l    = blockIdx.x * 64 + wave * 16;   // linear base pixel of this wave
  int n    = l / HWp;
  int rem  = l % HWp;
  int h    = rem / Ww;
  int w0   = rem % Ww;
  _Float16* frag = smem + wave * 3072;

  // ---- Phase 1: build val[K=192][N=16] f16 fragments in LDS ----
  const float* sb = src + (size_t)n * Cc * HWp;
  for (int p = lane; p < 48; p += 32) {      // 48 = 16 pixels * 3 taps
    int pixel = p & 15;
    int tap   = p >> 4;
    size_t ci = (((size_t)n * 3 + tap) * Hh + h) * Ww + (w0 + pixel);
    float py = PY[ci], px = PX[ci], m = M[ci];
    float y0f = floorf(py), x0f = floorf(px);
    int y0 = (int)y0f, x0 = (int)x0f;
    int y1 = y0 + 1,  x1 = x0 + 1;
    float wy = py - y0f, wx = px - x0f;
    bool vy0 = (y0 >= 0) & (y0 < Hh), vy1 = (y1 >= 0) & (y1 < Hh);
    bool vx0 = (x0 >= 0) & (x0 < Ww), vx1 = (x1 >= 0) & (x1 < Ww);
    float w00 = (1.f-wy)*(1.f-wx) * ((vy0 & vx0) ? m : 0.f);
    float w01 = (1.f-wy)*wx       * ((vy0 & vx1) ? m : 0.f);
    float w10 = wy*(1.f-wx)       * ((vy1 & vx0) ? m : 0.f);
    float w11 = wy*wx             * ((vy1 & vx1) ? m : 0.f);
    int y0c = y0 < 0 ? 0 : (y0 > Hh-1 ? Hh-1 : y0);
    int y1c = y1 < 0 ? 0 : (y1 > Hh-1 ? Hh-1 : y1);
    int x0c = x0 < 0 ? 0 : (x0 > Ww-1 ? Ww-1 : x0);
    int x1c = x1 < 0 ? 0 : (x1 > Ww-1 ? Ww-1 : x1);
    int o00 = y0c*Ww + x0c, o01 = y0c*Ww + x1c;
    int o10 = y1c*Ww + x0c, o11 = y1c*Ww + x1c;
    int Klin0 = tap * 64;
    for (int c = 0; c < Cc; ++c) {
      const float* sc = sb + (size_t)c * HWp;
      float v = w00*sc[o00] + w01*sc[o01] + w10*sc[o10] + w11*sc[o11];
      int K_lin = Klin0 + c;
      int t  = K_lin >> 5;
      int kk = K_lin & 31;
      int j  = ((kk & 16) >> 1) | (kk & 7);       // half index within v16h
      int fl = pixel + ((kk & 8) << 1);           // fragment lane
      frag[t * 512 + fl * 16 + j] = (_Float16)v;  // ds_store_b16
    }
  }
  __syncthreads();

  // ---- Phase 2: D = sum_t A(t) x B(t), 4 output-channel tiles ----
  v8f acc[4];
#pragma unroll
  for (int i = 0; i < 4; ++i)
#pragma unroll
    for (int r = 0; r < 8; ++r) acc[i][r] = 0.f;

  for (int t = 0; t < 6; ++t) {
    union { uint4 q[2]; v16h h; } bu;
    const uint4* bp = (const uint4*)(frag + t * 512 + lane * 16);
    bu.q[0] = bp[0];
    bu.q[1] = bp[1];
#pragma unroll
    for (int ot = 0; ot < 4; ++ot) {
      union { uint4 q[2]; v16h h; } au;
      const uint4* ap = (const uint4*)(wfrag + ((ot * 6 + t) * 32 + lane) * 16);
      au.q[0] = ap[0];
      au.q[1] = ap[1];
      acc[ot] = __builtin_amdgcn_wmma_f32_16x16x32_f16(
          false, au.h, false, bu.h, (short)0, acc[ot], false, false);
    }
  }

  // ---- Epilogue: C/D layout (lane<16: M=r, N=lane; lane>=16: M=r+8) ----
  int mrow = (lane >= 16) ? 8 : 0;
  int wpix = w0 + (lane & 15);
#pragma unroll
  for (int ot = 0; ot < 4; ++ot)
#pragma unroll
    for (int r = 0; r < 8; ++r) {
      int o = ot * 16 + r + mrow;
      dst[(((size_t)n * Cc + o) * Hh + h) * Ww + wpix] = acc[ot][r] + bias[o];
    }
}

// ---------------------------------------------------------------------------
extern "C" void kernel_launch(void* const* d_in, const int* in_sizes, int n_in,
                              void* d_out, int out_size, void* d_ws, size_t ws_size,
                              hipStream_t stream) {
  const float* x        = (const float*)d_in[0];
  const float* w_off_h  = (const float*)d_in[1];
  const float* b_off_h  = (const float*)d_in[2];
  const float* w_mask_h = (const float*)d_in[3];
  const float* b_mask_h = (const float*)d_in[4];
  const float* w_off_v  = (const float*)d_in[5];
  const float* b_off_v  = (const float*)d_in[6];
  const float* w_mask_v = (const float*)d_in[7];
  const float* b_mask_v = (const float*)d_in[8];
  const float* w_h      = (const float*)d_in[9];
  const float* b_h      = (const float*)d_in[10];
  const float* w_v      = (const float*)d_in[11];
  const float* b_v      = (const float*)d_in[12];
  float* out = (float*)d_out;

  char*  ws    = (char*)d_ws;
  float* xh    = (float*)ws;                                // 32 MiB intermediate
  float* coord = (float*)(ws + XH_ELEMS * 4);
  float* PYH = coord;           float* PXH = PYH + NKHW;  float* MH = PXH + NKHW;
  float* PYV = MH + NKHW;       float* PXV = PYV + NKHW;  float* MV = PXV + NKHW;
  _Float16* wfrag = (_Float16*)(ws + XH_ELEMS * 4 + 6 * NKHW * 4);  // 16B aligned

  prep_weights_kernel<<<(2 * WFRAG_HALFS + 255) / 256, 256, 0, stream>>>(
      w_h, w_v, wfrag);
  offsets_masks_kernel<<<(Bn * Hh * Ww + 255) / 256, 256, 0, stream>>>(
      x, w_off_h, b_off_h, w_mask_h, b_mask_h,
      w_off_v, b_off_v, w_mask_v, b_mask_v,
      PYH, PXH, MH, PYV, PXV, MV);
  // Pass 0 (horizontal): gather from x, write xh
  deform_pass_kernel<<<Bn * Hh * Ww / 64, 128, 0, stream>>>(
      x, PYH, PXH, MH, wfrag, b_h, xh);
  // Pass 1 (vertical): gather from xh, write final output
  deform_pass_kernel<<<Bn * Hh * Ww / 64, 128, 0, stream>>>(
      xh, PYV, PXV, MV, wfrag + WFRAG_HALFS, b_v, out);
}
